// CloudEncoder_46866683134155
// MI455X (gfx1250) — matrix operand order repaired and checked by reading, not compiled, not run on hardware
//
#include <hip/hip_runtime.h>
#include <hip/hip_bf16.h>

typedef __attribute__((ext_vector_type(16))) _Float16 v16h;
typedef __attribute__((ext_vector_type(8)))  _Float16 v8h;
typedef __attribute__((ext_vector_type(8)))  float    v8f;

#define FR   16    // total frames = B*F
#define SEQ  512   // points per frame
#define SP   528   // padded S1 (513 -> 528, multiple of 16)
#define SPK  544   // padded key dim for attn@V (multiple of 32)
#define DM   256
#define NH   8
#define DH   32
#define DFFC 1024
#define NL   4
#define KNN  50

union FragU { v16h v; v8h h[2]; };

struct GemmP {
  const _Float16* A; long aO, aI; int lda;
  const _Float16* B; long bO, bI; int ldb;
  float*       C;  long cO,  cI;  int ldc;
  _Float16*    Ch; long chO, chI; int ldch;
  const float* bias;                          // per-column bias (nullable)
  const float* add; long adO, adI; int ldad;  // full-matrix add (nullable)
  float scale; int relu;
  int M, N, Kd, inner;
};

// ---------------- generic batched WMMA GEMM: C = scale*(A @ B^T) + bias + add ----
__global__ __launch_bounds__(256) void gemm_wmma_f16(GemmP p) {
  int z  = blockIdx.z;
  int zo = z / p.inner, zi = z % p.inner;
  int wv   = threadIdx.x >> 5;
  int lane = threadIdx.x & 31;
  int l16  = lane & 15, hi = lane >> 4;
  int m0 = blockIdx.y * 128 + wv * 16;
  if (m0 >= p.M) return;                       // uniform per wave
  int n0 = blockIdx.x * 64;

  const _Float16* A = p.A + zo * p.aO + zi * p.aI;
  const _Float16* B = p.B + zo * p.bO + zi * p.bI;
  const _Float16* rowA = A + (long)(m0 + l16) * p.lda;
  const _Float16* rowB[4];
  bool ok[4];
#pragma unroll
  for (int j = 0; j < 4; ++j) {
    int n = n0 + 16 * j;
    ok[j] = (n < p.N);
    int nb = ok[j] ? n : 0;                    // clamp edge subtiles to valid rows;
    rowB[j] = B + (long)(nb + l16) * p.ldb;    // results discarded in epilogue
  }
  const v8f z8 = {0.f,0.f,0.f,0.f,0.f,0.f,0.f,0.f};
  v8f acc[4];
#pragma unroll
  for (int j = 0; j < 4; ++j) acc[j] = z8;

  // Branch-free hot loop: clause of b128 loads, then 4 independent WMMA chains.
#pragma unroll 2
  for (int k0 = 0; k0 < p.Kd; k0 += 32) {
    FragU a;
    // A 16x32 f16 fragment: lane half 'hi' holds K = hi*16..+15
    a.h[0] = *(const v8h*)(rowA + k0 + hi * 16);
    a.h[1] = *(const v8h*)(rowA + k0 + hi * 16 + 8);
    FragU b0, b1, b2, b3;
    // B 32x16 f16 fragment (B^T row-major): lane half 'hi' holds K = hi*16..+15
    b0.h[0] = *(const v8h*)(rowB[0] + k0 + hi * 16);
    b0.h[1] = *(const v8h*)(rowB[0] + k0 + hi * 16 + 8);
    b1.h[0] = *(const v8h*)(rowB[1] + k0 + hi * 16);
    b1.h[1] = *(const v8h*)(rowB[1] + k0 + hi * 16 + 8);
    b2.h[0] = *(const v8h*)(rowB[2] + k0 + hi * 16);
    b2.h[1] = *(const v8h*)(rowB[2] + k0 + hi * 16 + 8);
    b3.h[0] = *(const v8h*)(rowB[3] + k0 + hi * 16);
    b3.h[1] = *(const v8h*)(rowB[3] + k0 + hi * 16 + 8);
    acc[0] = __builtin_amdgcn_wmma_f32_16x16x32_f16(
        false, a.v, false, b0.v, (short)0, acc[0], false, false);
    acc[1] = __builtin_amdgcn_wmma_f32_16x16x32_f16(
        false, a.v, false, b1.v, (short)0, acc[1], false, false);
    acc[2] = __builtin_amdgcn_wmma_f32_16x16x32_f16(
        false, a.v, false, b2.v, (short)0, acc[2], false, false);
    acc[3] = __builtin_amdgcn_wmma_f32_16x16x32_f16(
        false, a.v, false, b3.v, (short)0, acc[3], false, false);
  }

  float*       Cp  = p.C  ? p.C  + zo * p.cO  + zi * p.cI  : (float*)0;
  _Float16*    Chp = p.Ch ? p.Ch + zo * p.chO + zi * p.chI : (_Float16*)0;
  const float* Ad  = p.add ? p.add + zo * p.adO + zi * p.adI : (const float*)0;
#pragma unroll
  for (int j = 0; j < 4; ++j) {
    if (!ok[j]) continue;
    int n = n0 + 16 * j + l16;
    float bcol = p.bias ? p.bias[n] : 0.f;
#pragma unroll
    for (int r = 0; r < 8; ++r) {
      int m = m0 + r + 8 * hi;                 // C layout: VGPR r -> row r (+8 for hi half)
      float v = acc[j][r] * p.scale + bcol;
      if (Ad) v += Ad[(long)m * p.ldad + n];
      if (p.relu) v = v > 0.f ? v : 0.f;
      if (Cp)  Cp [(long)m * p.ldc  + n] = v;
      if (Chp) Chp[(long)m * p.ldch + n] = (_Float16)v;
    }
  }
}

// ---------------- weight convert + transpose: WT[n][k] = (f16)W[k][n] ----------
__global__ void convT(const float* W, _Float16* WT, int Kr, int Nc) {
  long base = (long)blockIdx.y * Kr * Nc;
  long tot = (long)Kr * Nc;
  for (long i = (long)blockIdx.x * blockDim.x + threadIdx.x; i < tot;
       i += (long)gridDim.x * blockDim.x) {
    long n = i / Kr, k = i % Kr;
    WT[base + i] = (_Float16)W[base + k * Nc + n];
  }
}

// ---------------- positional encoding + query token + row padding --------------
__global__ __launch_bounds__(256) void pos_encode(const float* x, const float* qe,
                                                  float* srcF, _Float16* srcH) {
  int i = blockIdx.x, fr = blockIdx.y, c = threadIdx.x;
  float v;
  if (i > SEQ) {
    v = 0.f;
  } else if (i == SEQ) {
    v = qe[c];
  } else if (c < 4) {
    v = 0.f;                                   // left zero-pad (256 - 252)
  } else {
    int j = c - 4, axis = j / 84, jj = j % 84, e = jj >> 1;
    float xv = x[((long)fr * SEQ + i) * 3 + axis];
    const float bin = 0.002f / 0.015f;
    float q = floorf(xv / bin);
    float p = q / powf(10000.f, (float)e * (1.f / 42.f));
    v = (jj & 1) ? cosf(p) : sinf(p);
  }
  long o = ((long)fr * SP + i) * DM + c;
  srcF[o] = v; srcH[o] = (_Float16)v;
}

// ---------------- kNN mask -> additive attention bias --------------------------
__global__ __launch_bounds__(256) void knn_bias(const float* x, float* biasM) {
  int i = blockIdx.x, fr = blockIdx.y, t = threadIdx.x;
  float* row = biasM + ((long)fr * SP + i) * SP;
  if (i >= SEQ) {                              // query row + padded rows: keys 0..512 allowed
    for (int j = t; j < SP; j += 256) row[j] = (j <= SEQ) ? 0.f : -1e9f;
    return;
  }
  __shared__ float d2[SEQ];
  __shared__ unsigned long long red[256];
  const float* xb = x + (long)fr * SEQ * 3;
  float x0 = xb[i*3], x1 = xb[i*3+1], x2 = xb[i*3+2];
  for (int j = t; j < SEQ; j += 256) {
    float a = xb[j*3] - x0, b = xb[j*3+1] - x1, c = xb[j*3+2] - x2;
    d2[j] = a*a + b*b + c*c;
  }
  for (int j = t; j < SP; j += 256) row[j] = (j == SEQ) ? 0.f : -1e9f;
  __syncthreads();
  for (int it = 0; it < KNN; ++it) {           // 50x argmin, tie-break on lower index
    unsigned long long best = ~0ull;
    for (int j = t; j < SEQ; j += 256) {
      unsigned long long k = ((unsigned long long)__float_as_uint(d2[j]) << 32) | (unsigned)j;
      if (k < best) best = k;
    }
    red[t] = best; __syncthreads();
    for (int s = 128; s > 0; s >>= 1) {
      if (t < s) { if (red[t + s] < red[t]) red[t] = red[t + s]; }
      __syncthreads();
    }
    int idx = (int)(red[0] & 0xffffffffu);
    if (t == 0) { row[idx] = 0.f; d2[idx] = 3.4e38f; }
    __syncthreads();
  }
}

// ---------------- in-place fp16 row softmax (row length 528, pad to 544) -------
__global__ __launch_bounds__(256) void softmax_rows(_Float16* S) {
  int i = blockIdx.x, z = blockIdx.y, t = threadIdx.x;
  _Float16* row = S + ((long)z * SP + i) * SPK;
  float v0 = (float)row[t];
  float v1 = (float)row[t + 256];
  float v2 = (t + 512 < SP) ? (float)row[t + 512] : -__builtin_inff();
  __shared__ float red[256];
  float mx = fmaxf(fmaxf(v0, v1), v2);
  red[t] = mx; __syncthreads();
  for (int s = 128; s > 0; s >>= 1) { if (t < s) red[t] = fmaxf(red[t], red[t + s]); __syncthreads(); }
  mx = red[0]; __syncthreads();
  float e0 = __expf(v0 - mx), e1 = __expf(v1 - mx);
  float e2 = (t + 512 < SP) ? __expf(v2 - mx) : 0.f;
  red[t] = e0 + e1 + e2; __syncthreads();
  for (int s = 128; s > 0; s >>= 1) { if (t < s) red[t] += red[t + s]; __syncthreads(); }
  float rs = 1.f / red[0];
  row[t] = (_Float16)(e0 * rs);
  row[t + 256] = (_Float16)(e1 * rs);
  if (t + 512 < SP) row[t + 512] = (_Float16)(e2 * rs);
  if (t < SPK - SP) row[SP + t] = (_Float16)0.f;   // zero K-pad for attn@V
}

// ---------------- V transpose per frame: VT[c][s] = V[s][c], zero s-pad --------
__global__ __launch_bounds__(256) void transposeV(const _Float16* Vh, _Float16* VT) {
  int fr = blockIdx.y;
  long idx = (long)blockIdx.x * 256 + threadIdx.x;   // over DM*SPK
  int c = (int)(idx / SPK), s = (int)(idx % SPK);
  _Float16 v = (s < SP) ? Vh[((long)fr * SP + s) * DM + c] : (_Float16)0.f;
  VT[((long)fr * DM + c) * SPK + s] = v;
}

// ---------------- merge heads: Ah[i][c] = Ob[h][i][c%32] -----------------------
__global__ __launch_bounds__(256) void merge_heads(const _Float16* Ob, _Float16* Ah) {
  int i = blockIdx.x, fr = blockIdx.y, c = threadIdx.x;
  Ah[((long)fr * SP + i) * DM + c] =
      Ob[(((long)fr * NH + (c >> 5)) * SP + i) * DH + (c & 31)];
}

// ---------------- residual add + LayerNorm -------------------------------------
__global__ __launch_bounds__(256) void add_ln(float* srcF, _Float16* srcH,
                                              const float* tmp,
                                              const float* g, const float* b) {
  int i = blockIdx.x, fr = blockIdx.y, c = threadIdx.x;
  long o = ((long)fr * SP + i) * DM + c;
  float z = srcF[o] + tmp[o];
  __shared__ float rs[256], rq[256];
  rs[c] = z; rq[c] = z * z; __syncthreads();
  for (int s = 128; s > 0; s >>= 1) {
    if (c < s) { rs[c] += rs[c + s]; rq[c] += rq[c + s]; }
    __syncthreads();
  }
  float m = rs[0] * (1.f / 256.f);
  float var = rq[0] * (1.f / 256.f) - m * m;
  float r = rsqrtf(var + 1e-5f);
  float y = (z - m) * r * g[c] + b[c];
  srcF[o] = y; srcH[o] = (_Float16)y;
}

// ---------------- final output: query-token row per frame ----------------------
__global__ void write_out(const float* srcF, float* out) {
  int fr = blockIdx.x, c = threadIdx.x;
  out[(long)fr * DM + c] = srcF[((long)fr * SP + SEQ) * DM + c];
}

extern "C" void kernel_launch(void* const* d_in, const int* in_sizes, int n_in,
                              void* d_out, int out_size, void* d_ws, size_t ws_size,
                              hipStream_t stream) {
  (void)in_sizes; (void)n_in; (void)out_size; (void)ws_size;
  const float* x   = (const float*)d_in[0];
  // d_in[1] = is_pad (all false) -- intentionally unused
  const float* qe  = (const float*)d_in[2];
  const float* Wq  = (const float*)d_in[3];
  const float* bq  = (const float*)d_in[4];
  const float* Wk  = (const float*)d_in[5];
  const float* bk  = (const float*)d_in[6];
  const float* Wv  = (const float*)d_in[7];
  const float* bv  = (const float*)d_in[8];
  const float* Wo  = (const float*)d_in[9];
  const float* bo  = (const float*)d_in[10];
  const float* g1  = (const float*)d_in[11];
  const float* be1 = (const float*)d_in[12];
  const float* g2  = (const float*)d_in[13];
  const float* be2 = (const float*)d_in[14];
  const float* W1  = (const float*)d_in[15];
  const float* b1  = (const float*)d_in[16];
  const float* W2  = (const float*)d_in[17];
  const float* b2  = (const float*)d_in[18];

  char* base = (char*)d_ws;
  size_t off = 0;
  auto carve = [&](size_t bytes) -> char* {
    char* p = base + off;
    off += (bytes + 255) & ~(size_t)255;
    return p;
  };
  float*    srcF = (float*)   carve((size_t)FR * SP * DM * 4);
  _Float16* srcH = (_Float16*)carve((size_t)FR * SP * DM * 2);
  _Float16* Qh   = (_Float16*)carve((size_t)FR * SP * DM * 2);
  _Float16* Kh   = (_Float16*)carve((size_t)FR * SP * DM * 2);
  _Float16* Vh   = (_Float16*)carve((size_t)FR * SP * DM * 2);
  _Float16* VT   = (_Float16*)carve((size_t)FR * DM * SPK * 2);
  _Float16* Sc   = (_Float16*)carve((size_t)FR * NH * SP * SPK * 2);
  _Float16* Ob   = (_Float16*)carve((size_t)FR * NH * SP * DH * 2);
  _Float16* Ah   = (_Float16*)carve((size_t)FR * SP * DM * 2);
  _Float16* F1   = (_Float16*)carve((size_t)FR * SP * DFFC * 2);
  float*    tmpF = (float*)   carve((size_t)FR * SP * DM * 4);
  float*    biasM= (float*)   carve((size_t)FR * SP * SP * 4);
  _Float16* wqT  = (_Float16*)carve((size_t)NL * DM * DM * 2);
  _Float16* wkT  = (_Float16*)carve((size_t)NL * DM * DM * 2);
  _Float16* wvT  = (_Float16*)carve((size_t)NL * DM * DM * 2);
  _Float16* woT  = (_Float16*)carve((size_t)NL * DM * DM * 2);
  _Float16* w1T  = (_Float16*)carve((size_t)NL * DM * DFFC * 2);
  _Float16* w2T  = (_Float16*)carve((size_t)NL * DFFC * DM * 2);

  convT<<<dim3(256,  NL), 256, 0, stream>>>(Wq, wqT, DM, DM);
  convT<<<dim3(256,  NL), 256, 0, stream>>>(Wk, wkT, DM, DM);
  convT<<<dim3(256,  NL), 256, 0, stream>>>(Wv, wvT, DM, DM);
  convT<<<dim3(256,  NL), 256, 0, stream>>>(Wo, woT, DM, DM);
  convT<<<dim3(1024, NL), 256, 0, stream>>>(W1, w1T, DM, DFFC);
  convT<<<dim3(1024, NL), 256, 0, stream>>>(W2, w2T, DFFC, DM);

  pos_encode<<<dim3(SP, FR), 256, 0, stream>>>(x, qe, srcF, srcH);
  knn_bias  <<<dim3(SP, FR), 256, 0, stream>>>(x, biasM);

  const int GY = (SP + 127) / 128;   // 5
  for (int l = 0; l < NL; ++l) {
    // --- Q, K, V projections ---
    auto qkv = [&](const float* bias, _Float16* out, const _Float16* WT) {
      GemmP p{};
      p.A = srcH; p.aO = (long)SP * DM; p.aI = 0; p.lda = DM;
      p.B = WT + (long)l * DM * DM; p.bO = 0; p.bI = 0; p.ldb = DM;
      p.Ch = out; p.chO = (long)SP * DM; p.chI = 0; p.ldch = DM;
      p.bias = bias + (long)l * DM;
      p.scale = 1.f; p.M = SP; p.N = DM; p.Kd = DM; p.inner = 1;
      gemm_wmma_f16<<<dim3(DM / 64, GY, FR), 256, 0, stream>>>(p);
    };
    qkv(bq, Qh, wqT); qkv(bk, Kh, wkT); qkv(bv, Vh, wvT);

    transposeV<<<dim3((DM * SPK) / 256, FR), 256, 0, stream>>>(Vh, VT);

    // --- scores = scale * Q_h @ K_h^T + mask-bias (fp16 out) ---
    { GemmP p{};
      p.A = Qh; p.aO = (long)SP * DM; p.aI = DH; p.lda = DM;
      p.B = Kh; p.bO = (long)SP * DM; p.bI = DH; p.ldb = DM;
      p.Ch = Sc; p.chO = (long)NH * SP * SPK; p.chI = (long)SP * SPK; p.ldch = SPK;
      p.add = biasM; p.adO = (long)SP * SP; p.adI = 0; p.ldad = SP;
      p.scale = 0.17677669529663687f;
      p.M = SP; p.N = SP; p.Kd = DH; p.inner = NH;
      gemm_wmma_f16<<<dim3((SP + 63) / 64, GY, FR * NH), 256, 0, stream>>>(p);
    }
    softmax_rows<<<dim3(SP, FR * NH), 256, 0, stream>>>(Sc);

    // --- O_h = attn_h @ V_h (head-major fp16 out) ---
    { GemmP p{};
      p.A = Sc; p.aO = (long)NH * SP * SPK; p.aI = (long)SP * SPK; p.lda = SPK;
      p.B = VT; p.bO = (long)DM * SPK; p.bI = (long)DH * SPK; p.ldb = SPK;
      p.Ch = Ob; p.chO = (long)NH * SP * DH; p.chI = (long)SP * DH; p.ldch = DH;
      p.scale = 1.f; p.M = SP; p.N = DH; p.Kd = SPK; p.inner = NH;
      gemm_wmma_f16<<<dim3(1, GY, FR * NH), 256, 0, stream>>>(p);
    }
    merge_heads<<<dim3(SP, FR), 256, 0, stream>>>(Ob, Ah);

    // --- O @ Wo + bo -> tmpF ---
    { GemmP p{};
      p.A = Ah; p.aO = (long)SP * DM; p.lda = DM;
      p.B = woT + (long)l * DM * DM; p.ldb = DM;
      p.C = tmpF; p.cO = (long)SP * DM; p.ldc = DM;
      p.bias = bo + (long)l * DM;
      p.scale = 1.f; p.M = SP; p.N = DM; p.Kd = DM; p.inner = 1;
      gemm_wmma_f16<<<dim3(DM / 64, GY, FR), 256, 0, stream>>>(p);
    }
    add_ln<<<dim3(SP, FR), 256, 0, stream>>>(srcF, srcH, tmpF,
                                             g1 + (long)l * DM, be1 + (long)l * DM);

    // --- FF1: relu(src @ W1 + b1) -> F1 (fp16) ---
    { GemmP p{};
      p.A = srcH; p.aO = (long)SP * DM; p.lda = DM;
      p.B = w1T + (long)l * DM * DFFC; p.ldb = DM;
      p.Ch = F1; p.chO = (long)SP * DFFC; p.ldch = DFFC;
      p.bias = b1 + (long)l * DFFC; p.relu = 1;
      p.scale = 1.f; p.M = SP; p.N = DFFC; p.Kd = DM; p.inner = 1;
      gemm_wmma_f16<<<dim3(DFFC / 64, GY, FR), 256, 0, stream>>>(p);
    }
    // --- FF2: F1 @ W2 + b2 -> tmpF ---
    { GemmP p{};
      p.A = F1; p.aO = (long)SP * DFFC; p.lda = DFFC;
      p.B = w2T + (long)l * DFFC * DM; p.ldb = DFFC;
      p.C = tmpF; p.cO = (long)SP * DM; p.ldc = DM;
      p.bias = b2 + (long)l * DM;
      p.scale = 1.f; p.M = SP; p.N = DM; p.Kd = DFFC; p.inner = 1;
      gemm_wmma_f16<<<dim3(DM / 64, GY, FR), 256, 0, stream>>>(p);
    }
    add_ln<<<dim3(SP, FR), 256, 0, stream>>>(srcF, srcH, tmpF,
                                             g2 + (long)l * DM, be2 + (long)l * DM);
  }

  write_out<<<dim3(FR), 256, 0, stream>>>(srcF, (float*)d_out);
}
